// CustomFluxVDAttnProcessor2_0_26938034880745
// MI455X (gfx1250) — matrix operand order repaired and checked by reading, not compiled, not run on hardware
//
#include <hip/hip_runtime.h>
#include <stdint.h>

typedef __attribute__((ext_vector_type(16))) __bf16  v16bf;
typedef __attribute__((ext_vector_type(8)))  float   v8f;

#define DMODEL 3072
#define NHEAD  24
#define HD     128
#define S_IMG_ 2048
#define S_TXT_ 512
#define S_TOT_ 2560
#define A_TOK_ 4
#define A_DIM_ 1024

union Frag16 { uint4 q[2]; v16bf v; };

__device__ __forceinline__ unsigned short f2bf(float f) {
  unsigned u = __float_as_uint(f);
  u += 0x7FFFu + ((u >> 16) & 1u);        // round-to-nearest-even
  return (unsigned short)(u >> 16);
}
__device__ __forceinline__ float bf2f(unsigned short h) {
  return __uint_as_float(((unsigned)h) << 16);
}

// ---------------------------------------------------------------------------
// Tiled GEMM: C[M,N] = A[M,K] @ W[K,N] + bias   (fp32 I/O, bf16 WMMA compute)
// grid = (N/128, M/128), block = 256 (8 waves). K % 32 == 0, tiles exact.
// ---------------------------------------------------------------------------
#define GP 40  // LDS row stride in halfwords (32 data + 8 pad; 80B, 16B-aligned)

__launch_bounds__(256)
__global__ void gemm_bias_kernel(const float* __restrict__ A, int lda,
                                 const float* __restrict__ W, int ldw,
                                 const float* __restrict__ bias,
                                 float* __restrict__ C, int ldc, int K)
{
  __shared__ unsigned short As[128 * GP];   // A tile [m][k]  bf16
  __shared__ unsigned short Bs[128 * GP];   // W tile transposed [n][k] bf16

  const int tid  = threadIdx.x;
  const int lane = tid & 31;
  const int wave = tid >> 5;
  const int wm   = wave & 3;     // 4 wave rows * 32
  const int wn   = wave >> 2;    // 2 wave cols * 64
  const int lr   = lane & 15;
  const int hf   = lane >> 4;
  const int m0   = blockIdx.y * 128;
  const int n0   = blockIdx.x * 128;

  v8f acc[2][4] = {};

  const int arow = tid >> 1, akb = (tid & 1) * 16;       // A staging map
  const int wk0  = (tid >> 4) * 2, wn8 = (tid & 15) * 8; // W staging map

  for (int kb = 0; kb < K; kb += 32) {
    __syncthreads();
    { // stage A: fp32 -> bf16, row-major [m][k]
      const float* src = A + (size_t)(m0 + arow) * lda + kb + akb;
      unsigned* dst = (unsigned*)(As + arow * GP + akb);
#pragma unroll
      for (int i = 0; i < 8; ++i) {
        float2 f = ((const float2*)src)[i];
        dst[i] = (unsigned)f2bf(f.x) | ((unsigned)f2bf(f.y) << 16);
      }
      if (kb + 32 < K) __builtin_prefetch(src + 32, 0, 0);   // next A k-tile
    }
    { // stage W: fp32 -> bf16, transposed into [n][k] (k-pairs packed)
      const float* s0 = W + (size_t)(kb + wk0) * ldw + n0 + wn8;
      const float* s1 = s0 + ldw;
#pragma unroll
      for (int i = 0; i < 8; ++i) {
        unsigned v = (unsigned)f2bf(s0[i]) | ((unsigned)f2bf(s1[i]) << 16);
        *(unsigned*)(Bs + (wn8 + i) * GP + wk0) = v;
      }
      if (kb + 32 < K) __builtin_prefetch(s0 + (size_t)32 * ldw, 0, 0);  // next W k-tile
    }
    __syncthreads();

    Frag16 af[2], bf[4];
#pragma unroll
    for (int mb = 0; mb < 2; ++mb) {   // A layout: k = (v>>2)*16 + hf*8 + (v&3)*2
      const unsigned short* p = As + (wm * 32 + mb * 16 + lr) * GP;
      af[mb].q[0] = *(const uint4*)(p + hf * 8);
      af[mb].q[1] = *(const uint4*)(p + 16 + hf * 8);
    }
#pragma unroll
    for (int nb = 0; nb < 4; ++nb) {   // B layout: k = (v>>2)*16 + hf*8 + (v&3)*2
      const unsigned short* p = Bs + (wn * 64 + nb * 16 + lr) * GP + hf * 8;
      bf[nb].q[0] = *(const uint4*)p;
      bf[nb].q[1] = *(const uint4*)(p + 16);
    }
#pragma unroll
    for (int mb = 0; mb < 2; ++mb)
#pragma unroll
      for (int nb = 0; nb < 4; ++nb)
        acc[mb][nb] = __builtin_amdgcn_wmma_f32_16x16x32_bf16(
            false, af[mb].v, false, bf[nb].v, (short)0, acc[mb][nb], false, false);
  }

#pragma unroll
  for (int mb = 0; mb < 2; ++mb)
#pragma unroll
    for (int nb = 0; nb < 4; ++nb)
#pragma unroll
      for (int r = 0; r < 8; ++r) {
        int m = m0 + wm * 32 + mb * 16 + hf * 8 + r;
        int n = n0 + wn * 64 + nb * 16 + lr;
        C[(size_t)m * ldc + n] = acc[mb][nb][r] + bias[n];
      }
}

// ---------------------------------------------------------------------------
// Per-head RMSNorm (q,k) + RoPE (q,k) + bf16 scatter into [H][S_TOT][HD]
// grid = (S_TOT, NHEAD), block = 32 (one wave per (s,h) row; 4 dims/lane)
// ---------------------------------------------------------------------------
__launch_bounds__(32)
__global__ void rmsrope_kernel(const float* __restrict__ qi, const float* __restrict__ ki,
                               const float* __restrict__ vi,
                               const float* __restrict__ qt, const float* __restrict__ kt,
                               const float* __restrict__ vt,
                               const float* __restrict__ nqw, const float* __restrict__ nkw,
                               const float* __restrict__ naqw, const float* __restrict__ nakw,
                               const float* __restrict__ fcos, const float* __restrict__ fsin,
                               unsigned short* __restrict__ Qb, unsigned short* __restrict__ Kb,
                               unsigned short* __restrict__ Vb)
{
  const int s = blockIdx.x, h = blockIdx.y, t = threadIdx.x;
  const int d0 = t * 4;
  const float *qr, *kr, *vr, *qw, *kw;
  if (s < S_TXT_) {
    size_t r = s;
    qr = qt + r * DMODEL; kr = kt + r * DMODEL; vr = vt + r * DMODEL;
    qw = naqw; kw = nakw;
  } else {
    size_t r = s - S_TXT_;
    qr = qi + r * DMODEL; kr = ki + r * DMODEL; vr = vi + r * DMODEL;
    qw = nqw; kw = nkw;
  }
  const int off = h * HD + d0;
  float qv[4], kv[4], vv[4];
#pragma unroll
  for (int i = 0; i < 4; ++i) { qv[i] = qr[off + i]; kv[i] = kr[off + i]; vv[i] = vr[off + i]; }
  float sq = 0.f, sk = 0.f;
#pragma unroll
  for (int i = 0; i < 4; ++i) { sq += qv[i] * qv[i]; sk += kv[i] * kv[i]; }
  for (int m = 16; m >= 1; m >>= 1) { sq += __shfl_xor(sq, m, 32); sk += __shfl_xor(sk, m, 32); }
  float rq = rsqrtf(sq * (1.f / HD) + 1e-6f);
  float rk = rsqrtf(sk * (1.f / HD) + 1e-6f);
#pragma unroll
  for (int i = 0; i < 4; ++i) { qv[i] *= rq * qw[d0 + i]; kv[i] *= rk * kw[d0 + i]; }
#pragma unroll
  for (int p = 0; p < 4; p += 2) {  // rope pairs (d0+p, d0+p+1)
    float c0 = fcos[(size_t)s * HD + d0 + p],     s0 = fsin[(size_t)s * HD + d0 + p];
    float c1 = fcos[(size_t)s * HD + d0 + p + 1], s1 = fsin[(size_t)s * HD + d0 + p + 1];
    float x1 = qv[p], x2 = qv[p + 1];
    qv[p] = x1 * c0 - x2 * s0; qv[p + 1] = x2 * c1 + x1 * s1;
    x1 = kv[p]; x2 = kv[p + 1];
    kv[p] = x1 * c0 - x2 * s0; kv[p + 1] = x2 * c1 + x1 * s1;
  }
  size_t ob = ((size_t)h * S_TOT_ + s) * HD + d0;
#pragma unroll
  for (int p = 0; p < 4; p += 2) {
    ((unsigned*)(Qb + ob))[p >> 1] = (unsigned)f2bf(qv[p]) | ((unsigned)f2bf(qv[p + 1]) << 16);
    ((unsigned*)(Kb + ob))[p >> 1] = (unsigned)f2bf(kv[p]) | ((unsigned)f2bf(kv[p + 1]) << 16);
    ((unsigned*)(Vb + ob))[p >> 1] = (unsigned)f2bf(vv[p]) | ((unsigned)f2bf(vv[p + 1]) << 16);
  }
}

// ---------------------------------------------------------------------------
// Flash attention over the joint sequence. grid = (S_TOT/128, NHEAD), block 256.
// Each wave owns 16 q-rows (full softmax row in-wave). 64-key tiles.
// K tile staged with CDNA5 async global->LDS DMA (ASYNCcnt).
// ---------------------------------------------------------------------------
#define KSP 136  // Ks row stride over d   (128+8 halfs, 272B)
#define VSP 72   // Vs row stride over key (64+8 halfs, 144B)
#define PSP 72   // Ps row stride over key

__launch_bounds__(256)
__global__ void flash_attn_kernel(const unsigned short* __restrict__ Qb,
                                  const unsigned short* __restrict__ Kb,
                                  const unsigned short* __restrict__ Vb,
                                  float* __restrict__ hs)
{
  __shared__ unsigned short Ks[64 * KSP];       // K tile [key][d]
  __shared__ unsigned short Vs[HD * VSP];       // V tile transposed [d][key]
  __shared__ unsigned short Ps[8 * 16 * PSP];   // per-wave P [m][key]

  const int h   = blockIdx.y;
  const int q0  = blockIdx.x * 128;
  const int tid = threadIdx.x, lane = tid & 31, wave = tid >> 5;
  const int lr  = lane & 15, hf = lane >> 4;

  // load Q fragments once (A layout), 16 q-rows per wave, K=128 -> 4 ksteps
  Frag16 qf[4];
  {
    const unsigned short* qrow = Qb + ((size_t)h * S_TOT_ + q0 + wave * 16 + lr) * HD;
#pragma unroll
    for (int kk = 0; kk < 4; ++kk) {
      qf[kk].q[0] = *(const uint4*)(qrow + kk * 32 + hf * 8);
      qf[kk].q[1] = *(const uint4*)(qrow + kk * 32 + 16 + hf * 8);
    }
  }

  v8f of[8] = {};
  float mst[8], lst[8];
#pragma unroll
  for (int r = 0; r < 8; ++r) { mst[r] = -1e30f; lst[r] = 0.f; }
  const float sc = 0.08838834764831845f;  // 1/sqrt(128)

  const int krow = tid >> 2, kseg = (tid & 3) * 32;
  // per-thread LDS byte address for async K staging (addr[31:0] = LDS offset)
  const unsigned kdst = (unsigned)(uintptr_t)(Ks + krow * KSP + kseg);

  for (int t0 = 0; t0 < S_TOT_; t0 += 64) {
    __syncthreads();
    { // stage K tile: async global->LDS copy (bf16, no register round-trip)
      const unsigned short* gsrc = Kb + ((size_t)h * S_TOT_ + t0 + krow) * HD + kseg;
#pragma unroll
      for (int i = 0; i < 4; ++i) {
        unsigned long long ga = (unsigned long long)(uintptr_t)(gsrc + i * 8);
        unsigned la = kdst + i * 16;
        asm volatile("global_load_async_to_lds_b128 %0, %1, off"
                     :: "v"(la), "v"(ga) : "memory");
      }
    }
    { // stage V tile transposed: Vs[d][key] (needs registers for transpose)
      const uint4* src = (const uint4*)(Vb + ((size_t)h * S_TOT_ + t0 + krow) * HD + kseg);
      uint4 tmp[4];
#pragma unroll
      for (int i = 0; i < 4; ++i) tmp[i] = src[i];
      const unsigned short* u = (const unsigned short*)tmp;
#pragma unroll
      for (int i = 0; i < 32; ++i) Vs[(kseg + i) * VSP + krow] = u[i];
    }
    asm volatile("s_wait_asynccnt 0" ::: "memory");  // this wave's async K writes landed
    __syncthreads();                                 // all waves' writes visible

    // S = Q @ K^T : 16 q-rows x 64 keys per wave
    v8f sf[4] = {};
#pragma unroll
    for (int j = 0; j < 4; ++j)
#pragma unroll
      for (int kk = 0; kk < 4; ++kk) {
        Frag16 b;  // B layout: n = key, k = d; k = (v>>2)*16 + hf*8 + (v&3)*2
        const unsigned short* p = Ks + (j * 16 + lr) * KSP + kk * 32 + hf * 8;
        b.q[0] = *(const uint4*)p;
        b.q[1] = *(const uint4*)(p + 16);
        sf[j] = __builtin_amdgcn_wmma_f32_16x16x32_bf16(
            false, qf[kk].v, false, b.v, (short)0, sf[j], false, false);
      }

    // online softmax (row m = hf*8 + r lives in one 16-lane half)
    unsigned short* pw = Ps + wave * (16 * PSP);
#pragma unroll
    for (int r = 0; r < 8; ++r) {
      float mx = -1e30f;
#pragma unroll
      for (int j = 0; j < 4; ++j) { sf[j][r] *= sc; mx = fmaxf(mx, sf[j][r]); }
      for (int m = 8; m >= 1; m >>= 1) mx = fmaxf(mx, __shfl_xor(mx, m, 32));
      float mn = fmaxf(mst[r], mx);
      float corr = __expf(mst[r] - mn);
      mst[r] = mn;
      float rs = 0.f;
#pragma unroll
      for (int j = 0; j < 4; ++j) { float p = __expf(sf[j][r] - mn); sf[j][r] = p; rs += p; }
      for (int m = 8; m >= 1; m >>= 1) rs += __shfl_xor(rs, m, 32);
      lst[r] = lst[r] * corr + rs;
#pragma unroll
      for (int nb = 0; nb < 8; ++nb) of[nb][r] *= corr;
#pragma unroll
      for (int j = 0; j < 4; ++j)   // C-layout -> LDS [m][key]
        pw[(hf * 8 + r) * PSP + j * 16 + lr] = f2bf(sf[j][r]);
    }

    // O += P @ V   (K = 64 keys -> 2 ksteps)
#pragma unroll
    for (int kk = 0; kk < 2; ++kk) {
      Frag16 pf;  // A layout: m = lr, k = key
      const unsigned short* pp = pw + lr * PSP + kk * 32;
      pf.q[0] = *(const uint4*)(pp + hf * 8);
      pf.q[1] = *(const uint4*)(pp + 16 + hf * 8);
#pragma unroll
      for (int nb = 0; nb < 8; ++nb) {
        Frag16 vf;  // B layout: n = d, k = key
        const unsigned short* vp = Vs + (nb * 16 + lr) * VSP + kk * 32 + hf * 8;
        vf.q[0] = *(const uint4*)vp;
        vf.q[1] = *(const uint4*)(vp + 16);
        of[nb] = __builtin_amdgcn_wmma_f32_16x16x32_bf16(
            false, pf.v, false, vf.v, (short)0, of[nb], false, false);
      }
    }
  }

  // write hs[s][h*HD + d]
#pragma unroll
  for (int nb = 0; nb < 8; ++nb)
#pragma unroll
    for (int r = 0; r < 8; ++r) {
      int s = q0 + wave * 16 + hf * 8 + r;
      int d = nb * 16 + lr;
      hs[(size_t)s * DMODEL + h * HD + d] = of[nb][r] / lst[r];
    }
}

// ---------------------------------------------------------------------------
// Adapter K/V projection: [4,1024] @ [1024,3072] (no bias), trivial size
// ---------------------------------------------------------------------------
__launch_bounds__(256)
__global__ void adapter_kv_kernel(const float* __restrict__ ae,
                                  const float* __restrict__ Wk, const float* __restrict__ Wv,
                                  float* __restrict__ vdK, float* __restrict__ vdV)
{
  int n = blockIdx.x * 256 + threadIdx.x;
  float aK[4] = {0.f, 0.f, 0.f, 0.f}, aV[4] = {0.f, 0.f, 0.f, 0.f};
  for (int k = 0; k < A_DIM_; ++k) {
    float wk = Wk[(size_t)k * DMODEL + n];
    float wv = Wv[(size_t)k * DMODEL + n];
#pragma unroll
    for (int t = 0; t < 4; ++t) {
      float a = ae[t * A_DIM_ + k];
      aK[t] += a * wk; aV[t] += a * wv;
    }
  }
#pragma unroll
  for (int t = 0; t < 4; ++t) { vdK[(size_t)t * DMODEL + n] = aK[t]; vdV[(size_t)t * DMODEL + n] = aV[t]; }
}

// ---------------------------------------------------------------------------
// Adapter cross-attention (4 keys) added into hs. grid=(S_TOT,NHEAD), block=32.
// ---------------------------------------------------------------------------
__launch_bounds__(32)
__global__ void vd_attn_kernel(const unsigned short* __restrict__ Qb,
                               const float* __restrict__ vdK, const float* __restrict__ vdV,
                               const float* __restrict__ bscale,
                               float* __restrict__ hs)
{
  const int s = blockIdx.x, h = blockIdx.y, t = threadIdx.x;
  const int d0 = t * 4;
  float qv[4];
  const unsigned short* qrow = Qb + ((size_t)h * S_TOT_ + s) * HD + d0;
#pragma unroll
  for (int i = 0; i < 4; ++i) qv[i] = bf2f(qrow[i]);
  const float scl = 0.08838834764831845f;
  float sj[4];
#pragma unroll
  for (int j = 0; j < 4; ++j) {
    float a = 0.f;
#pragma unroll
    for (int i = 0; i < 4; ++i) a += qv[i] * vdK[(size_t)j * DMODEL + h * HD + d0 + i];
    for (int m = 16; m >= 1; m >>= 1) a += __shfl_xor(a, m, 32);
    sj[j] = a * scl;
  }
  float mx = fmaxf(fmaxf(sj[0], sj[1]), fmaxf(sj[2], sj[3]));
  float e[4], se = 0.f;
#pragma unroll
  for (int j = 0; j < 4; ++j) { e[j] = __expf(sj[j] - mx); se += e[j]; }
  float inv = 1.f / se;
  float bs = bscale[0];  // SCALE == 1.0
  float* hr = hs + (size_t)s * DMODEL + h * HD + d0;
#pragma unroll
  for (int i = 0; i < 4; ++i) {
    float o = 0.f;
#pragma unroll
    for (int j = 0; j < 4; ++j) o += e[j] * inv * vdV[(size_t)j * DMODEL + h * HD + d0 + i];
    hr[i] += o * bs;
  }
}

// ---------------------------------------------------------------------------
extern "C" void kernel_launch(void* const* d_in, const int* in_sizes, int n_in,
                              void* d_out, int out_size, void* d_ws, size_t ws_size,
                              hipStream_t stream)
{
  (void)in_sizes; (void)n_in; (void)out_size; (void)ws_size;
  const float* hidden = (const float*)d_in[0];
  const float* enc    = (const float*)d_in[1];
  const float* ae     = (const float*)d_in[2];
  const float* fcos   = (const float*)d_in[3];
  const float* fsin   = (const float*)d_in[4];
  const float* Wq  = (const float*)d_in[5];  const float* bq  = (const float*)d_in[6];
  const float* Wk  = (const float*)d_in[7];  const float* bk  = (const float*)d_in[8];
  const float* Wv  = (const float*)d_in[9];  const float* bv  = (const float*)d_in[10];
  const float* nqw = (const float*)d_in[11]; const float* nkw = (const float*)d_in[12];
  const float* Wqa = (const float*)d_in[13]; const float* bqa = (const float*)d_in[14];
  const float* Wka = (const float*)d_in[15]; const float* bka = (const float*)d_in[16];
  const float* Wva = (const float*)d_in[17]; const float* bva = (const float*)d_in[18];
  const float* naqw = (const float*)d_in[19]; const float* nakw = (const float*)d_in[20];
  const float* Wo  = (const float*)d_in[21]; const float* bo  = (const float*)d_in[22];
  const float* Woa = (const float*)d_in[23]; const float* boa = (const float*)d_in[24];
  const float* Wkad = (const float*)d_in[25]; const float* Wvad = (const float*)d_in[26];
  const float* bsc  = (const float*)d_in[27];

  char* p = (char*)d_ws;
  auto alloc = [&](size_t bytes) { void* r = (void*)p; p += (bytes + 255) & ~(size_t)255; return r; };
  float* qi = (float*)alloc((size_t)S_IMG_ * DMODEL * 4);
  float* ki = (float*)alloc((size_t)S_IMG_ * DMODEL * 4);
  float* vi = (float*)alloc((size_t)S_IMG_ * DMODEL * 4);
  float* qt = (float*)alloc((size_t)S_TXT_ * DMODEL * 4);
  float* kt = (float*)alloc((size_t)S_TXT_ * DMODEL * 4);
  float* vt = (float*)alloc((size_t)S_TXT_ * DMODEL * 4);
  unsigned short* Qb = (unsigned short*)alloc((size_t)NHEAD * S_TOT_ * HD * 2);
  unsigned short* Kb = (unsigned short*)alloc((size_t)NHEAD * S_TOT_ * HD * 2);
  unsigned short* Vb = (unsigned short*)alloc((size_t)NHEAD * S_TOT_ * HD * 2);
  float* hs  = (float*)alloc((size_t)S_TOT_ * DMODEL * 4);
  float* vdK = (float*)alloc((size_t)A_TOK_ * DMODEL * 4);
  float* vdV = (float*)alloc((size_t)A_TOK_ * DMODEL * 4);

  dim3 b256(256);
  // QKV projections (image + text)
  gemm_bias_kernel<<<dim3(24, 16), b256, 0, stream>>>(hidden, DMODEL, Wq, DMODEL, bq, qi, DMODEL, DMODEL);
  gemm_bias_kernel<<<dim3(24, 16), b256, 0, stream>>>(hidden, DMODEL, Wk, DMODEL, bk, ki, DMODEL, DMODEL);
  gemm_bias_kernel<<<dim3(24, 16), b256, 0, stream>>>(hidden, DMODEL, Wv, DMODEL, bv, vi, DMODEL, DMODEL);
  gemm_bias_kernel<<<dim3(24, 4),  b256, 0, stream>>>(enc, DMODEL, Wqa, DMODEL, bqa, qt, DMODEL, DMODEL);
  gemm_bias_kernel<<<dim3(24, 4),  b256, 0, stream>>>(enc, DMODEL, Wka, DMODEL, bka, kt, DMODEL, DMODEL);
  gemm_bias_kernel<<<dim3(24, 4),  b256, 0, stream>>>(enc, DMODEL, Wva, DMODEL, bva, vt, DMODEL, DMODEL);
  // RMSNorm + RoPE + scatter to head-major bf16
  rmsrope_kernel<<<dim3(S_TOT_, NHEAD), 32, 0, stream>>>(qi, ki, vi, qt, kt, vt,
                                                         nqw, nkw, naqw, nakw,
                                                         fcos, fsin, Qb, Kb, Vb);
  // main joint flash attention -> hs
  flash_attn_kernel<<<dim3(S_TOT_ / 128, NHEAD), b256, 0, stream>>>(Qb, Kb, Vb, hs);
  // adapter branch
  adapter_kv_kernel<<<dim3(DMODEL / 256), b256, 0, stream>>>(ae, Wkad, Wvad, vdK, vdV);
  vd_attn_kernel<<<dim3(S_TOT_, NHEAD), 32, 0, stream>>>(Qb, vdK, vdV, bsc, hs);
  // output projections directly into d_out (img first, then enc)
  float* out_img = (float*)d_out;
  float* out_enc = out_img + (size_t)S_IMG_ * DMODEL;
  gemm_bias_kernel<<<dim3(24, 16), b256, 0, stream>>>(hs + (size_t)S_TXT_ * DMODEL, DMODEL, Wo, DMODEL, bo, out_img, DMODEL, DMODEL);
  gemm_bias_kernel<<<dim3(24, 4),  b256, 0, stream>>>(hs, DMODEL, Woa, DMODEL, boa, out_enc, DMODEL, DMODEL);
}
